// ConvOffset2d_42417097016719
// MI455X (gfx1250) — matrix operand order repaired and checked by reading, not compiled, not run on hardware
//
#include <hip/hip_runtime.h>
#include <math.h>

// ---------------- problem constants ----------------
constexpr int Bn   = 8;
constexpr int CIN  = 256;
constexpr int Hh   = 64;
constexpr int Wd   = 64;
constexpr int COUT = 256;
constexpr int KK   = 9;        // 3x3 taps
constexpr int HO   = 64;
constexpr int WO   = 64;

constexpr int NT = 64;         // N tile (spatial positions) per block
constexpr int KC = 32;         // K chunk (bf16 WMMA K)
constexpr int KCHUNKS = (CIN * KK) / KC;   // 72
constexpr int LDS_STRIDE = 36; // halves per row (32 payload + 4 pad) = 72 bytes

typedef __attribute__((ext_vector_type(16))) __bf16 v16bf;
typedef __attribute__((ext_vector_type(8)))  float  v8f;
typedef __attribute__((ext_vector_type(4)))  unsigned int u32x4;
typedef __attribute__((ext_vector_type(8)))  int i32x8;
typedef __attribute__((ext_vector_type(4)))  int i32x4;

#if defined(__has_builtin)
#if __has_builtin(__builtin_amdgcn_tensor_load_to_lds) && \
    __has_builtin(__builtin_amdgcn_s_wait_tensorcnt)
#define HAS_TDM 1
#endif
#endif
#ifndef HAS_TDM
#define HAS_TDM 0
#endif

union FragU {
    v16bf v;
    uint2 q[4];
};

__device__ __forceinline__ unsigned short f2bf(float f) {
    union { float f; unsigned int u; } v;
    v.f = f;
    unsigned int r = v.u + 0x7FFFu + ((v.u >> 16) & 1u);   // round to nearest even
    return (unsigned short)(r >> 16);
}

// ---------------- NCHW -> NHWC transpose (per batch: [C][HW] -> [HW][C]) ----
__global__ __launch_bounds__(256) void nchw_to_nhwc(const float* __restrict__ in,
                                                    float* __restrict__ out) {
    __shared__ float tile[32][33];
    const int b   = blockIdx.z;
    const int hw0 = blockIdx.x * 32;
    const int c0  = blockIdx.y * 32;
    const int tx  = threadIdx.x;
    const int ty  = threadIdx.y;

    const float* pin = in + (size_t)b * CIN * (Hh * Wd);
#pragma unroll
    for (int j = ty; j < 32; j += 8)
        tile[j][tx] = pin[(size_t)(c0 + j) * (Hh * Wd) + hw0 + tx];
    __syncthreads();
    float* pout = out + (size_t)b * (Hh * Wd) * CIN;
#pragma unroll
    for (int j = ty; j < 32; j += 8)
        pout[(size_t)(hw0 + j) * CIN + c0 + tx] = tile[tx][j];
}

// ---- weight repack: [COUT][CIN][3][3] f32 -> [72][COUT][32] bf16 (chunk-major)
__global__ __launch_bounds__(256) void pack_weights(const float* __restrict__ w,
                                                    unsigned short* __restrict__ wp) {
    const int idx = blockIdx.x * 256 + threadIdx.x;  // [kk][o][kc]
    const int kc  = idx & 31;
    const int o   = (idx >> 5) & 255;
    const int kk  = idx >> 13;
    const int tap = kk >> 3;
    const int c   = (kk & 7) * KC + kc;
    wp[idx] = f2bf(w[((size_t)o * CIN + c) * KK + tap]);
}

// =======================================================================
// Main fused kernel: TDM-staged weights (double buffered) + bilinear im2col
// + bf16 WMMA.  One M pass: 8 waves x 32 output channels = 256.
// =======================================================================
__global__ __launch_bounds__(256) void deform_conv_wmma_tdm(
    const float* __restrict__ xT,           // [B][H][W][C] (transposed x)
    const float* __restrict__ offs,         // [B][18][HO][WO]
    const unsigned short* __restrict__ wp,  // [72][256][32] bf16
    float* __restrict__ out)                // [B][COUT][HO][WO]
{
    __shared__ __align__(16) unsigned short Alds[2][COUT * LDS_STRIDE];
    __shared__ __align__(16) unsigned short Blds[NT * LDS_STRIDE];
    __shared__ int   smpY[NT * KK];
    __shared__ int   smpX[NT * KK];
    __shared__ float smpWy[NT * KK];
    __shared__ float smpWx[NT * KK];

    const int tid  = threadIdx.x;
    const int lane = tid & 31;
    const int wave = tid >> 5;             // 8 waves, each owns 32 M rows

    const int n0 = blockIdx.x * NT;        // tile shares (b,h); w = 0..63
    const int b  = n0 >> 12;
    const int h  = (n0 >> 6) & 63;

    // ---- per-(position, tap) bilinear metadata ----
    for (int idx = tid; idx < NT * KK; idx += 256) {
        const int nl  = idx & 63;
        const int tap = idx >> 6;
        const int ky  = tap / 3;
        const int kx  = tap - ky * 3;
        const size_t ob = (((size_t)b * (2 * KK) + tap * 2) * HO + h) * WO + nl;
        const float dy = offs[ob];
        const float dx = offs[ob + HO * WO];
        const float py = (float)(h - 1 + ky) + dy;
        const float px = (float)(nl - 1 + kx) + dx;
        const float fy = floorf(py);
        const float fx = floorf(px);
        smpY[idx]  = (int)fy;
        smpX[idx]  = (int)fx;
        smpWy[idx] = py - fy;
        smpWx[idx] = px - fx;
    }

#if HAS_TDM
    // TDM: 1-D DMA of one 16KB weight chunk into LDS, padding 2 DWORDs after
    // every 16 DWORDs -> 72-byte row stride in LDS (matches LDS_STRIDE).
    auto tdm_issue = [&](int kk, int buf) {
        const unsigned lds_lo =
            (unsigned)(size_t)&Alds[buf][0];          // low 32b = LDS byte addr
        const unsigned long long ga =
            (unsigned long long)(size_t)wp + (unsigned long long)kk * 16384ull;
        u32x4 g0 = { 1u,                              // count=1, user descriptor
                     lds_lo,
                     (unsigned)ga,
                     (unsigned)(ga >> 32) | 0x80000000u };  // type=2 ("image")
        // data_size=8B | pad_enable | pad_interval=16DW | pad_amount=2DW
        i32x8 g1 = { 0x02D30000,
                     0x08000000,                      // tensor_dim0 = 2048 (8B units)
                     0x00010000,                      // tensor_dim1 = 1
                     0x08000000,                      // tile_dim0 = 2048
                     0,                               // tile_dim1/2 = 0
                     2048,                            // tensor_dim0_stride
                     0, 0 };
        i32x4 gz4 = { 0, 0, 0, 0 };
        i32x8 gz8 = { 0, 0, 0, 0, 0, 0, 0, 0 };
        __builtin_amdgcn_tensor_load_to_lds(g0, g1, gz4, gz4, gz8, 0);
    };
    if (wave == 0) tdm_issue(0, 0);
#endif

    v8f acc[2][4] = {};

    for (int kk = 0; kk < KCHUNKS; ++kk) {
        const int buf   = kk & 1;
        const int tap   = kk >> 3;
        const int cbase = (kk & 7) * KC;

        __syncthreads();   // previous compute done: Blds / other A buf reusable

        // ---- stage B: bilinear im2col tile [NT][KC] -> bf16 LDS ----
        {
            const int nl = tid >> 2;              // 0..63
            const int cq = (tid & 3) * 8;         // 0,8,16,24
            const int s  = tap * 64 + nl;
            const int y0 = smpY[s], x0 = smpX[s];
            const float wy = smpWy[s], wx = smpWx[s];
            float v[8];
#pragma unroll
            for (int j = 0; j < 8; ++j) v[j] = 0.f;
            const float cw[4] = { (1.f - wy) * (1.f - wx), (1.f - wy) * wx,
                                  wy * (1.f - wx),         wy * wx };
            const int cy[4] = { y0, y0, y0 + 1, y0 + 1 };
            const int cx[4] = { x0, x0 + 1, x0, x0 + 1 };
#pragma unroll
            for (int c4 = 0; c4 < 4; ++c4) {
                const int yy = cy[c4], xx = cx[c4];
                if (yy >= 0 && yy < Hh && xx >= 0 && xx < Wd) {
                    const float w4 = cw[c4];
                    const float4* p = (const float4*)(xT +
                        ((((size_t)b * Hh + yy) * Wd + xx) * CIN + cbase + cq));
                    const float4 p0 = p[0], p1 = p[1];
                    v[0] += w4 * p0.x; v[1] += w4 * p0.y;
                    v[2] += w4 * p0.z; v[3] += w4 * p0.w;
                    v[4] += w4 * p1.x; v[5] += w4 * p1.y;
                    v[6] += w4 * p1.z; v[7] += w4 * p1.w;
                }
            }
            unsigned short* bp = &Blds[nl * LDS_STRIDE + cq];
#pragma unroll
            for (int j = 0; j < 8; ++j) bp[j] = f2bf(v[j]);
        }

#if HAS_TDM
        if (wave == 0) __builtin_amdgcn_s_wait_tensorcnt((short)0);
#else
        // fallback: cooperative copy of this chunk's weights into LDS
        {
            const unsigned short* s = wp + (size_t)kk * (COUT * KC) + tid * KC;
            unsigned short* d = &Alds[buf][tid * LDS_STRIDE];
#pragma unroll
            for (int j = 0; j < KC; ++j) d[j] = s[j];
        }
#endif
        __syncthreads();   // B visible everywhere; A(kk) DMA complete

#if HAS_TDM
        if (wave == 0 && kk + 1 < KCHUNKS) tdm_issue(kk + 1, buf ^ 1);
#endif

        // ---- fragments per documented CDNA5 layouts ----
        // A 16x32: lanes 0-15 K={0..7,16..23}, lanes 16-31 K={8..15,24..31}
        FragU afrag[2];
#pragma unroll
        for (int mi = 0; mi < 2; ++mi) {
            const int row = wave * 32 + mi * 16 + (lane & 15);
            const int kh  = (lane >> 4) * 8;
            const uint2* p0 = (const uint2*)&Alds[buf][row * LDS_STRIDE + kh];
            const uint2* p1 = (const uint2*)&Alds[buf][row * LDS_STRIDE + kh + 16];
            afrag[mi].q[0] = p0[0]; afrag[mi].q[1] = p0[1];
            afrag[mi].q[2] = p1[0]; afrag[mi].q[3] = p1[1];
        }
        // B 32x16: N=lane&15, lanes 0-15 K={0..7,16..23}, lanes 16-31 K={8..15,24..31}
#pragma unroll
        for (int ni = 0; ni < 4; ++ni) {
            const int col = ni * 16 + (lane & 15);
            const int kb  = (lane >> 4) * 8;
            const uint2* p0 = (const uint2*)&Blds[col * LDS_STRIDE + kb];
            const uint2* p1 = (const uint2*)&Blds[col * LDS_STRIDE + kb + 16];
            FragU bfrag;
            bfrag.q[0] = p0[0]; bfrag.q[1] = p0[1];
            bfrag.q[2] = p1[0]; bfrag.q[3] = p1[1];
#pragma unroll
            for (int mi = 0; mi < 2; ++mi)
                acc[mi][ni] = __builtin_amdgcn_wmma_f32_16x16x32_bf16(
                    false, afrag[mi].v, false, bfrag.v,
                    (short)0, acc[mi][ni], false, false);
        }
    }

    // ---- epilogue: C/D layout -> out[b][o][h][w] ----
#pragma unroll
    for (int mi = 0; mi < 2; ++mi) {
#pragma unroll
        for (int ni = 0; ni < 4; ++ni) {
            const int o  = wave * 32 + mi * 16 + (lane >> 4) * 8;
            const int nl = ni * 16 + (lane & 15);
            const size_t base = ((size_t)b * COUT + o) * (HO * WO) + h * WO + nl;
#pragma unroll
            for (int r = 0; r < 8; ++r)
                out[base + (size_t)r * (HO * WO)] = acc[mi][ni][r];
        }
    }
}

// =======================================================================
// Legacy fully-self-contained fallback (no workspace): NCHW reads, f32 weights
// =======================================================================
__global__ __launch_bounds__(256) void deform_conv_wmma_legacy(
    const float* __restrict__ xsrc, const float* __restrict__ offs,
    const float* __restrict__ wgt, float* __restrict__ out)
{
    constexpr int MT = 128;
    __shared__ __align__(16) unsigned short Alds[MT * LDS_STRIDE];
    __shared__ __align__(16) unsigned short Blds[NT * LDS_STRIDE];
    __shared__ int   smpY[NT * KK];
    __shared__ int   smpX[NT * KK];
    __shared__ float smpWy[NT * KK];
    __shared__ float smpWx[NT * KK];

    const int tid  = threadIdx.x;
    const int lane = tid & 31;
    const int wave = tid >> 5;
    const int wm   = wave & 3;
    const int wn   = wave >> 2;

    const int n0    = blockIdx.x * NT;
    const int Mbase = blockIdx.y * MT;
    const int b     = n0 >> 12;
    const int h     = (n0 >> 6) & 63;

    for (int idx = tid; idx < NT * KK; idx += 256) {
        const int nl  = idx & 63;
        const int tap = idx >> 6;
        const int ky  = tap / 3;
        const int kx  = tap - ky * 3;
        const size_t ob = (((size_t)b * (2 * KK) + tap * 2) * HO + h) * WO + nl;
        const float py = (float)(h - 1 + ky) + offs[ob];
        const float px = (float)(nl - 1 + kx) + offs[ob + HO * WO];
        const float fy = floorf(py);
        const float fx = floorf(px);
        smpY[idx]  = (int)fy;
        smpX[idx]  = (int)fx;
        smpWy[idx] = py - fy;
        smpWx[idx] = px - fx;
    }

    v8f acc[2][2] = {};

    for (int kk = 0; kk < KCHUNKS; ++kk) {
        const int tap   = kk >> 3;
        const int cbase = (kk & 7) * KC;
        __syncthreads();
        {
            const int ol = tid >> 1;
            const int ch = (tid & 1) * 16;
            const float* wp =
                wgt + ((size_t)(Mbase + ol) * CIN + (cbase + ch)) * KK + tap;
            unsigned short* ap = &Alds[ol * LDS_STRIDE + ch];
#pragma unroll
            for (int j = 0; j < 16; ++j) ap[j] = f2bf(wp[(size_t)j * KK]);
        }
        {
            const int nl = tid >> 2;
            const int cq = (tid & 3) * 8;
            const int s  = tap * 64 + nl;
            const int y0 = smpY[s], x0 = smpX[s];
            const float wy = smpWy[s], wx = smpWx[s];
            float v[8];
#pragma unroll
            for (int j = 0; j < 8; ++j) v[j] = 0.f;
            const float cw[4] = { (1.f - wy) * (1.f - wx), (1.f - wy) * wx,
                                  wy * (1.f - wx),         wy * wx };
            const int cy[4] = { y0, y0, y0 + 1, y0 + 1 };
            const int cx[4] = { x0, x0 + 1, x0, x0 + 1 };
#pragma unroll
            for (int c4 = 0; c4 < 4; ++c4) {
                const int yy = cy[c4], xx = cx[c4];
                if (yy >= 0 && yy < Hh && xx >= 0 && xx < Wd) {
                    const float w4 = cw[c4];
                    const float* p = xsrc +
                        ((size_t)b * CIN + cbase + cq) * (Hh * Wd) + yy * Wd + xx;
#pragma unroll
                    for (int j = 0; j < 8; ++j)
                        v[j] += w4 * p[(size_t)j * (Hh * Wd)];
                }
            }
            unsigned short* bp = &Blds[nl * LDS_STRIDE + cq];
#pragma unroll
            for (int j = 0; j < 8; ++j) bp[j] = f2bf(v[j]);
        }
        __syncthreads();

        FragU afrag[2], bfrag[2];
#pragma unroll
        for (int mi = 0; mi < 2; ++mi) {
            const int row = wm * 32 + mi * 16 + (lane & 15);
            const int kh  = (lane >> 4) * 8;
            const uint2* p0 = (const uint2*)&Alds[row * LDS_STRIDE + kh];
            const uint2* p1 = (const uint2*)&Alds[row * LDS_STRIDE + kh + 16];
            afrag[mi].q[0] = p0[0]; afrag[mi].q[1] = p0[1];
            afrag[mi].q[2] = p1[0]; afrag[mi].q[3] = p1[1];
        }
#pragma unroll
        for (int ni = 0; ni < 2; ++ni) {
            const int col = wn * 32 + ni * 16 + (lane & 15);
            const int kb  = (lane >> 4) * 8;
            const uint2* p0 = (const uint2*)&Blds[col * LDS_STRIDE + kb];
            const uint2* p1 = (const uint2*)&Blds[col * LDS_STRIDE + kb + 16];
            bfrag[ni].q[0] = p0[0]; bfrag[ni].q[1] = p0[1];
            bfrag[ni].q[2] = p1[0]; bfrag[ni].q[3] = p1[1];
        }
#pragma unroll
        for (int mi = 0; mi < 2; ++mi)
#pragma unroll
            for (int ni = 0; ni < 2; ++ni)
                acc[mi][ni] = __builtin_amdgcn_wmma_f32_16x16x32_bf16(
                    false, afrag[mi].v, false, bfrag[ni].v,
                    (short)0, acc[mi][ni], false, false);
    }

#pragma unroll
    for (int mi = 0; mi < 2; ++mi) {
#pragma unroll
        for (int ni = 0; ni < 2; ++ni) {
            const int o  = Mbase + wm * 32 + mi * 16 + (lane >> 4) * 8;
            const int nl = wn * 32 + ni * 16 + (lane & 15);
            const size_t base = ((size_t)b * COUT + o) * (HO * WO) + h * WO + nl;
#pragma unroll
            for (int r = 0; r < 8; ++r)
                out[base + (size_t)r * (HO * WO)] = acc[mi][ni][r];
        }
    }
}

extern "C" void kernel_launch(void* const* d_in, const int* in_sizes, int n_in,
                              void* d_out, int out_size, void* d_ws, size_t ws_size,
                              hipStream_t stream) {
    const float* x    = (const float*)d_in[0];
    const float* offs = (const float*)d_in[1];
    const float* wgt  = (const float*)d_in[2];
    float* out = (float*)d_out;

    const size_t xbytes  = (size_t)Bn * CIN * Hh * Wd * sizeof(float);      // 33.5MB
    const size_t wpbytes = (size_t)KCHUNKS * COUT * KC * sizeof(unsigned short);

    if (ws_size >= xbytes + wpbytes) {
        float* xT = (float*)d_ws;
        unsigned short* wp = (unsigned short*)((char*)d_ws + xbytes);

        dim3 gt((Hh * Wd) / 32, CIN / 32, Bn);
        dim3 bt(32, 8, 1);
        nchw_to_nhwc<<<gt, bt, 0, stream>>>(x, xT);
        pack_weights<<<(KCHUNKS * COUT * KC) / 256, 256, 0, stream>>>(wgt, wp);

        deform_conv_wmma_tdm<<<dim3((Bn * HO * WO) / NT, 1, 1), 256, 0, stream>>>(
            xT, offs, wp, out);
    } else {
        deform_conv_wmma_legacy<<<dim3((Bn * HO * WO) / NT, COUT / 128, 1),
                                  256, 0, stream>>>(x, offs, wgt, out);
    }
}
